// AutoregressiveDecoder_82360292868657
// MI455X (gfx1250) — matrix operand order repaired and checked by reading, not compiled, not run on hardware
//
#include <hip/hip_runtime.h>

// ---------------------------------------------------------------------------
// Autoregressive LSTM decoder for MI455X (gfx1250, wave32, WMMA).
// B=32, L=256, H=512, E=512, V=32000, G=4H=2048. T derived from out_size.
//
// Strategy: convert all GEMM weights to f16 once per launch, pre-packed into
// the exact v_wmma_f32_16x16x32_f16 per-lane operand layout, so the hot
// logits GEMM ([32,512]x[512,32000] per step, weights L2-resident at 32MB)
// issues v_wmma + b128 loads only. 64 sequential steps as a host loop of
// 4 stream-ordered kernels (graph capturable).
// ---------------------------------------------------------------------------

typedef __attribute__((ext_vector_type(16))) _Float16 v16h;
typedef __attribute__((ext_vector_type(8)))  float    v8f;

constexpr int B  = 32;
constexpr int L  = 256;
constexpr int H  = 512;
constexpr int E  = 512;
constexpr int V  = 32000;
constexpr int G  = 4 * H;        // 2048
constexpr int KT = 512 / 32;     // 16 k-tiles of K=32 (E == H == 512)

__device__ inline v8f wmma16(v16h a, v16h b, v8f c) {
  // D = A(16x32 f16) x B(32x16 f16) + C(16x16 f32)
  return __builtin_amdgcn_wmma_f32_16x16x32_f16(
      /*neg_a=*/false, a, /*neg_b=*/false, b,
      /*c_mod=*/(short)0, c, /*reuse_a=*/false, /*reuse_b=*/false);
}

// A-operand packed index for element (m, k) of a [M x 512] f16 matrix.
// ISA 16-bit A 16x32 layout: lane = (k-bit3 ? 16:0) + m%16,
// half j = 2*v + p with v = (k-bit4 ? 4:0) + k[2:1], p = k[0].
__device__ inline int packA_index(int m, int k) {
  int mt = m >> 4;
  int kt = k >> 5;
  int kin = k & 31;
  int a  = (kin >> 4) & 1;
  int bb = (kin >> 3) & 1;
  int cc = (kin >> 1) & 3;
  int p  = kin & 1;
  int lane = bb * 16 + (m & 15);
  int j = 2 * (a * 4 + cc) + p;
  return ((mt * KT + kt) << 9) + (lane << 4) + j;
}

// ---------------------------------------------------------------------------
// Pack a row-major [N x 512] f32 weight into f16 WMMA B-operand tiles.
// B 32x16 layout: lanes 0-15 hold K=0..7,16..23 (col = lane), lanes 16-31 hold
// K=8..15,24..31 (col = lane-16); half j = (k-bit4 ? 8:0) + k[2:0].
// ---------------------------------------------------------------------------
__global__ void pack_w_kernel(const float* __restrict__ W,
                              _Float16* __restrict__ out, int ntiles) {
  long tid = (long)blockIdx.x * blockDim.x + threadIdx.x;
  long total = (long)ntiles * KT * 512;
  if (tid >= total) return;
  int tile = (int)(tid >> 9);
  int e    = (int)(tid & 511);
  int nt = tile / KT;
  int kt = tile % KT;
  int lane = e >> 4;
  int j    = e & 15;
  int n = nt * 16 + (lane & 15);
  int k = kt * 32 + ((j >> 3) << 4) + ((lane >> 4) << 3) + (j & 7);
  out[tid] = (_Float16)W[(size_t)n * 512 + k];
}

// ---------------------------------------------------------------------------
// h0 = z @ Wh.T + bh ; c0 = z @ Wc.T + bc  (tiny, scalar FMA, float4 loads)
// Writes c (f32) and pack-writes h0 into both h_p and x_p (x = h0 at t=0).
// ---------------------------------------------------------------------------
__global__ void init_kernel(const float* __restrict__ z,
                            const float* __restrict__ Wh, const float* __restrict__ bh,
                            const float* __restrict__ Wc, const float* __restrict__ bc,
                            float* __restrict__ c,
                            _Float16* __restrict__ h_p, _Float16* __restrict__ x_p) {
  int tid = blockIdx.x * blockDim.x + threadIdx.x;   // B*H = 16384
  int b  = tid / H;
  int hi = tid % H;
  const float4* zr = (const float4*)(z + (size_t)b * L);
  const float4* wh = (const float4*)(Wh + (size_t)hi * L);
  const float4* wc = (const float4*)(Wc + (size_t)hi * L);
  float sh = bh[hi];
  float sc = bc[hi];
  for (int i = 0; i < L / 4; ++i) {
    float4 zv = zr[i];
    float4 hv = wh[i];
    float4 cv = wc[i];
    sh += zv.x * hv.x + zv.y * hv.y + zv.z * hv.z + zv.w * hv.w;
    sc += zv.x * cv.x + zv.y * cv.y + zv.z * cv.z + zv.w * cv.w;
  }
  c[tid] = sc;
  int pidx = packA_index(b, hi);
  _Float16 hf = (_Float16)sh;
  h_p[pidx] = hf;
  x_p[pidx] = hf;
}

// ---------------------------------------------------------------------------
// gates[32,2048] = x @ Wih.T + h @ Whh.T (bias added in cell kernel).
// 32 blocks x 4 waves; each wave owns one 16-col N tile, 2 M tiles, K=512.
// ---------------------------------------------------------------------------
__global__ void __launch_bounds__(128) gates_kernel(
    const _Float16* __restrict__ x_p, const _Float16* __restrict__ h_p,
    const _Float16* __restrict__ wih_p, const _Float16* __restrict__ whh_p,
    float* __restrict__ gates) {
  int lane = threadIdx.x & 31;
  int wave = threadIdx.x >> 5;
  int nt = blockIdx.x * 4 + wave;              // 0..127
  v8f acc0 = {};
  v8f acc1 = {};
  int lo = lane << 4;
  for (int kt = 0; kt < KT; ++kt) {
    v16h ax0 = *(const v16h*)(x_p + ((0 * KT + kt) << 9) + lo);
    v16h ax1 = *(const v16h*)(x_p + ((1 * KT + kt) << 9) + lo);
    v16h ah0 = *(const v16h*)(h_p + ((0 * KT + kt) << 9) + lo);
    v16h ah1 = *(const v16h*)(h_p + ((1 * KT + kt) << 9) + lo);
    v16h bx  = *(const v16h*)(wih_p + (((size_t)nt * KT + kt) << 9) + lo);
    v16h bh  = *(const v16h*)(whh_p + (((size_t)nt * KT + kt) << 9) + lo);
    acc0 = wmma16(ax0, bx, acc0);
    acc0 = wmma16(ah0, bh, acc0);
    acc1 = wmma16(ax1, bx, acc1);
    acc1 = wmma16(ah1, bh, acc1);
  }
  // C layout: lanes 0-15 -> col=lane, rows r; lanes 16-31 -> col=lane-16, rows r+8
  int col   = nt * 16 + (lane & 15);
  int rbase = (lane >> 4) * 8;
  for (int r = 0; r < 8; ++r) {
    gates[(size_t)(rbase + r) * G + col]      = acc0[r];
    gates[(size_t)(16 + rbase + r) * G + col] = acc1[r];
  }
}

// ---------------------------------------------------------------------------
// LSTM cell update; writes new c (f32) and packs h_new into h_p (A layout).
// ---------------------------------------------------------------------------
__global__ void cell_kernel(const float* __restrict__ gates,
                            const float* __restrict__ bih, const float* __restrict__ bhh,
                            float* __restrict__ c, _Float16* __restrict__ h_p) {
  int tid = blockIdx.x * blockDim.x + threadIdx.x;   // B*H
  int b  = tid / H;
  int hi = tid % H;
  size_t row = (size_t)b * G;
  float gi = gates[row + hi]         + bih[hi]         + bhh[hi];
  float gf = gates[row + H + hi]     + bih[H + hi]     + bhh[H + hi];
  float gg = gates[row + 2 * H + hi] + bih[2 * H + hi] + bhh[2 * H + hi];
  float go = gates[row + 3 * H + hi] + bih[3 * H + hi] + bhh[3 * H + hi];
  float si = 1.0f / (1.0f + __expf(-gi));
  float sf = 1.0f / (1.0f + __expf(-gf));
  float so = 1.0f / (1.0f + __expf(-go));
  float cn = sf * c[tid] + si * tanhf(gg);
  float hn = so * tanhf(cn);
  c[tid] = cn;
  h_p[packA_index(b, hi)] = (_Float16)hn;
}

// ---------------------------------------------------------------------------
// logits[32, 32000] = h_new @ Wfc.T + bfc, written to d_out[b, t, :].
// 250 blocks x 8 waves; A (32KB packed h) staged in LDS, shared by 8 waves.
// Wfc f16 (32MB) streams from L2.
// ---------------------------------------------------------------------------
__global__ void __launch_bounds__(256) logits_kernel(
    const _Float16* __restrict__ h_p, const _Float16* __restrict__ wfc_p,
    const float* __restrict__ bfc, float* __restrict__ out, int t, int T) {
  __shared__ __align__(32) _Float16 lds_a[2 * KT * 512];   // 32 KB
  {
    const uint4* src = (const uint4*)h_p;
    uint4* dst = (uint4*)lds_a;
    for (int i = threadIdx.x; i < (2 * KT * 512) / 8; i += 256) dst[i] = src[i];
  }
  __syncthreads();

  int lane = threadIdx.x & 31;
  int wave = threadIdx.x >> 5;
  int nt = blockIdx.x * 8 + wave;              // 0..1999
  int lo = lane << 4;
  float bias = bfc[nt * 16 + (lane & 15)];
  v8f acc0, acc1;
  for (int r = 0; r < 8; ++r) { acc0[r] = bias; acc1[r] = bias; }

  for (int kt = 0; kt < KT; ++kt) {
    v16h a0 = *(const v16h*)(lds_a + (kt << 9) + lo);
    v16h a1 = *(const v16h*)(lds_a + ((KT + kt) << 9) + lo);
    v16h bw = *(const v16h*)(wfc_p + (((size_t)nt * KT + kt) << 9) + lo);
    acc0 = wmma16(a0, bw, acc0);
    acc1 = wmma16(a1, bw, acc1);
  }

  int col   = nt * 16 + (lane & 15);
  int rbase = (lane >> 4) * 8;
  for (int r = 0; r < 8; ++r) {
    out[((size_t)(rbase + r) * T + t) * V + col]      = acc0[r];
    out[((size_t)(16 + rbase + r) * T + t) * V + col] = acc1[r];
  }
}

// ---------------------------------------------------------------------------
// Per-row argmax over V=32000 logits, gather embedding[pred], pack into x_p.
// One block per batch row.
// ---------------------------------------------------------------------------
__global__ void __launch_bounds__(256) argmax_embed_kernel(
    const float* __restrict__ out, const float* __restrict__ embedding,
    _Float16* __restrict__ x_p, int t, int T) {
  __shared__ float sv[256];
  __shared__ int   si[256];
  int b = blockIdx.x;
  const float* row = out + ((size_t)b * T + t) * V;
  float best = -__builtin_inff();
  int   bidx = 0;
  for (int v = threadIdx.x; v < V; v += 256) {
    float x = row[v];
    if (x > best) { best = x; bidx = v; }
  }
  sv[threadIdx.x] = best;
  si[threadIdx.x] = bidx;
  __syncthreads();
  for (int s = 128; s > 0; s >>= 1) {
    if (threadIdx.x < s) {
      float ov = sv[threadIdx.x + s];
      int   oi = si[threadIdx.x + s];
      if (ov > sv[threadIdx.x] ||
          (ov == sv[threadIdx.x] && oi < si[threadIdx.x])) {
        sv[threadIdx.x] = ov;
        si[threadIdx.x] = oi;
      }
    }
    __syncthreads();
  }
  int pred = si[0];
  const float* erow = embedding + (size_t)pred * E;
  for (int k = threadIdx.x; k < E; k += 256)
    x_p[packA_index(b, k)] = (_Float16)erow[k];
}

// ---------------------------------------------------------------------------
extern "C" void kernel_launch(void* const* d_in, const int* in_sizes, int n_in,
                              void* d_out, int out_size, void* d_ws, size_t ws_size,
                              hipStream_t stream) {
  const float* z         = (const float*)d_in[0];
  const float* embedding = (const float*)d_in[1];
  const float* Wh        = (const float*)d_in[2];
  const float* bh        = (const float*)d_in[3];
  const float* Wc        = (const float*)d_in[4];
  const float* bc        = (const float*)d_in[5];
  const float* Wih       = (const float*)d_in[6];
  const float* Whh       = (const float*)d_in[7];
  const float* bih       = (const float*)d_in[8];
  const float* bhh       = (const float*)d_in[9];
  const float* Wfc       = (const float*)d_in[10];
  const float* bfc       = (const float*)d_in[11];
  float* out = (float*)d_out;
  int T = out_size / (B * V);   // 64

  // Workspace carve-up (256B aligned).
  char* ws = (char*)d_ws;
  size_t o = 0;
  auto alloc = [&](size_t n) -> char* {
    o = (o + 255) & ~(size_t)255;
    char* p = ws + o;
    o += n;
    return p;
  };
  const int NT_G = G / 16;      // 128
  const int NT_V = V / 16;      // 2000
  _Float16* wih_p = (_Float16*)alloc((size_t)NT_G * KT * 512 * 2);  // 2 MB
  _Float16* whh_p = (_Float16*)alloc((size_t)NT_G * KT * 512 * 2);  // 2 MB
  _Float16* wfc_p = (_Float16*)alloc((size_t)NT_V * KT * 512 * 2);  // 32 MB
  float*    cbuf  = (float*)   alloc((size_t)B * H * 4);
  _Float16* x_p   = (_Float16*)alloc((size_t)2 * KT * 512 * 2);
  _Float16* h_p   = (_Float16*)alloc((size_t)2 * KT * 512 * 2);
  float*    gates = (float*)   alloc((size_t)B * G * 4);
  (void)ws_size; (void)in_sizes; (void)n_in;

  // One-time (per launch) weight conversion + packing.
  {
    long n1 = (long)NT_G * KT * 512;
    pack_w_kernel<<<(int)((n1 + 255) / 256), 256, 0, stream>>>(Wih, wih_p, NT_G);
    pack_w_kernel<<<(int)((n1 + 255) / 256), 256, 0, stream>>>(Whh, whh_p, NT_G);
    long n2 = (long)NT_V * KT * 512;
    pack_w_kernel<<<(int)((n2 + 255) / 256), 256, 0, stream>>>(Wfc, wfc_p, NT_V);
  }
  init_kernel<<<(B * H) / 256, 256, 0, stream>>>(z, Wh, bh, Wc, bc, cbuf, h_p, x_p);

  // Sequential autoregressive loop: 4 stream-ordered kernels per step.
  for (int t = 0; t < T; ++t) {
    gates_kernel<<<NT_G / 4, 128, 0, stream>>>(x_p, h_p, wih_p, whh_p, gates);
    cell_kernel<<<(B * H) / 256, 256, 0, stream>>>(gates, bih, bhh, cbuf, h_p);
    logits_kernel<<<NT_V / 8, 256, 0, stream>>>(h_p, wfc_p, bfc, out, t, T);
    argmax_embed_kernel<<<B, 256, 0, stream>>>(out, embedding, x_p, t, T);
  }
}
